// DynamicGraphConv_62577673503634
// MI455X (gfx1250) — matrix operand order repaired and checked by reading, not compiled, not run on hardware
//
#include <hip/hip_runtime.h>
#include <hip/hip_bf16.h>

// B=64, N=1024, H=256, OUT=256. Exploits:
//  - softmax adjacency row-sums == 1 (adaptive branch == identity)
//  - _conv only needs adjacency row-sums s0[n], s1[n]
//  - out[b,n,o] = acc0 + s0*acc1 + s0^2*acc2 + s1*acc3 + s1^2*acc4 + bias
//    where accj = x[b,n,:] @ Wj  (W035 = W0+W5+W6 folded into acc0 slice)

typedef __attribute__((ext_vector_type(16))) __bf16 v16bf;
typedef __attribute__((ext_vector_type(8)))  float  v8f;
typedef __attribute__((ext_vector_type(8)))  float  f32x8;

#define NN 1024
#define HH 256
#define OUTDIM 256

// ---------- prep 1: adjacency row sums ----------
__global__ __launch_bounds__(256) void rowsum_kernel(const float* __restrict__ sup0,
                                                     const float* __restrict__ sup1,
                                                     float* __restrict__ s0,
                                                     float* __restrict__ s1) {
    int row = blockIdx.x;                 // 0..2047: first 1024 -> s0, rest -> s1
    const float* src = (row < NN) ? (sup0 + (size_t)row * NN)
                                  : (sup1 + (size_t)(row - NN) * NN);
    float sum = 0.f;
    for (int k = threadIdx.x; k < NN; k += 256) sum += src[k];
    __shared__ float red[256];
    red[threadIdx.x] = sum;
    __syncthreads();
    for (int s = 128; s > 0; s >>= 1) {
        if (threadIdx.x < s) red[threadIdx.x] += red[threadIdx.x + s];
        __syncthreads();
    }
    if (threadIdx.x == 0) {
        float* dst = (row < NN) ? s0 : s1;
        dst[row & (NN - 1)] = red[0];
    }
}

// ---------- prep 2: fold W slices, transpose to [j][o][k], convert to bf16 ----------
__global__ __launch_bounds__(256) void wprep_kernel(const float* __restrict__ W,
                                                    __bf16* __restrict__ Wb) {
    int idx = blockIdx.x * 256 + threadIdx.x;     // 5*256*256 = 327680
    if (idx >= 5 * HH * OUTDIM) return;
    int j   = idx >> 16;          // 0..4
    int rem = idx & 65535;
    int o   = rem >> 8;           // 0..255
    int k   = rem & 255;          // 0..255
    float v;
    if (j == 0) {
        v = W[(size_t)k * OUTDIM + o]
          + W[(size_t)(5 * HH + k) * OUTDIM + o]
          + W[(size_t)(6 * HH + k) * OUTDIM + o];
    } else {
        v = W[(size_t)(j * HH + k) * OUTDIM + o];
    }
    Wb[idx] = (__bf16)v;          // layout: ((j*256 + o)*256 + k), k contiguous
}

// ---------- main GEMM: [65536,256] x 5x[256,256] with fused epilogue ----------
__global__ __launch_bounds__(256) void dgc_gemm_kernel(const float* __restrict__ x,
                                                       const __bf16* __restrict__ Wb,
                                                       const float* __restrict__ s0,
                                                       const float* __restrict__ s1,
                                                       const float* __restrict__ bias,
                                                       float* __restrict__ out) {
    const int lane = threadIdx.x & 31;
    const int wave = threadIdx.x >> 5;            // 0..7
    const int rowBase = blockIdx.x * 128 + wave * 16;   // wave's 16-row M tile
    const int oBase   = blockIdx.y * 32;                // wave's 32-col N range (2 tiles)

    const int  mrow   = lane & 15;                // A: lane -> matrix row M
    const bool hiHalf = (lane >= 16);
    const int  kA = hiHalf ? 16 : 0;              // A: lanes>=16 hold K 16..31
    const int  kB = hiHalf ? 16 : 0;              // B: lanes>=16 hold K 16..31

    const float* xrow = x + (size_t)(rowBase + mrow) * HH;

    v8f acc[5][2];
#pragma unroll
    for (int j = 0; j < 5; ++j)
#pragma unroll
        for (int t = 0; t < 2; ++t)
            acc[j][t] = (v8f){0.f, 0.f, 0.f, 0.f, 0.f, 0.f, 0.f, 0.f};

#pragma unroll
    for (int k0 = 0; k0 < HH; k0 += 32) {
        // --- A fragment: 16 f32 -> bf16, contiguous K (k = 16*half + e), same order as B ---
        f32x8 a0 = *(const f32x8*)(xrow + k0 + kA);        // K = k0+kA .. +7
        f32x8 a1 = *(const f32x8*)(xrow + k0 + kA + 8);    // K = k0+kA+8 .. +15
        v16bf afrag;
#pragma unroll
        for (int e = 0; e < 8; ++e) {
            afrag[e]     = (__bf16)a0[e];
            afrag[e + 8] = (__bf16)a1[e];
        }
#pragma unroll
        for (int j = 0; j < 5; ++j) {
#pragma unroll
            for (int t = 0; t < 2; ++t) {
                // B fragment: column o, contiguous K (32 bytes)
                const __bf16* bp = Wb
                    + ((size_t)(j * OUTDIM + (oBase + 16 * t + (lane & 15)))) * HH
                    + k0 + kB;
                v16bf bfrag = *(const v16bf*)bp;
                acc[j][t] = __builtin_amdgcn_wmma_f32_16x16x32_bf16(
                    false, afrag, false, bfrag, (short)0, acc[j][t], false, false);
            }
        }
    }

    // --- epilogue: per-row coefficient combine + bias, C/D layout:
    //     row = v + (lane>=16 ? 8 : 0), col = lane & 15 ---
    const int rLaneBase = rowBase + (hiHalf ? 8 : 0);
#pragma unroll
    for (int t = 0; t < 2; ++t) {
        const int o = oBase + 16 * t + (lane & 15);
        const float bo = bias[o];
#pragma unroll
        for (int v = 0; v < 8; ++v) {
            const int r = rLaneBase + v;
            const int n = r & (NN - 1);           // r = b*1024 + n
            const float c1 = s0[n];
            const float c3 = s1[n];
            const float val = acc[0][t][v]
                            + c1 * acc[1][t][v]
                            + (c1 * c1) * acc[2][t][v]
                            + c3 * acc[3][t][v]
                            + (c3 * c3) * acc[4][t][v]
                            + bo;
            out[(size_t)r * OUTDIM + o] = val;
        }
    }
}

extern "C" void kernel_launch(void* const* d_in, const int* in_sizes, int n_in,
                              void* d_out, int out_size, void* d_ws, size_t ws_size,
                              hipStream_t stream) {
    const float* x        = (const float*)d_in[0];   // [64,1024,256]
    const float* support0 = (const float*)d_in[1];   // [1024,1024]
    const float* support1 = (const float*)d_in[2];   // [1024,1024]
    // d_in[3], d_in[4] (node_vec1/2) are mathematically unused: softmax row-sums == 1
    const float* W        = (const float*)d_in[5];   // [1792,256]
    const float* bias     = (const float*)d_in[6];   // [256]
    float* out = (float*)d_out;                      // [64,1024,256]

    // workspace layout
    float*  s0 = (float*)d_ws;                               // 1024 f32
    float*  s1 = s0 + NN;                                    // 1024 f32
    __bf16* Wb = (__bf16*)((char*)d_ws + 8192);              // 5*256*256 bf16

    rowsum_kernel<<<2 * NN, 256, 0, stream>>>(support0, support1, s0, s1);
    wprep_kernel<<<(5 * HH * OUTDIM) / 256, 256, 0, stream>>>(W, Wb);

    dim3 grid(65536 / 128, OUTDIM / 32);   // (512, 8)
    dgc_gemm_kernel<<<grid, 256, 0, stream>>>(x, Wb, s0, s1, bias, out);
}
